// GenericRNNSequenceEncoder_87342454932146
// MI455X (gfx1250) — matrix operand order repaired and checked by reading, not compiled, not run on hardware
//
#include <hip/hip_runtime.h>
#include <hip/hip_bf16.h>
#include <stdint.h>

// ---------------------------------------------------------------------------
// LSTM sequence encoder for MI455X (gfx1250, wave32, WMMA + TDM).
// Phase 1: x_proj = X @ Wi + b   (bf16 WMMA GEMM, f32 accumulate)
// Phase 2: persistent 16-WG scan; Wh pinned in VGPR B-fragments; per-step
//          x_proj tile prefetched into LDS by the Tensor Data Mover.
// ---------------------------------------------------------------------------

typedef __attribute__((ext_vector_type(16))) __bf16 v16bf;
typedef __attribute__((ext_vector_type(8)))  __bf16 v8bf;
typedef __attribute__((ext_vector_type(4)))  __bf16 v4bf;
typedef __attribute__((ext_vector_type(8)))  float  v8f;

typedef __attribute__((ext_vector_type(4))) unsigned int tdm_g0_t;
typedef __attribute__((ext_vector_type(8))) int          tdm_g1_t;
typedef __attribute__((ext_vector_type(4))) int          tdm_g2_t;
typedef __attribute__((ext_vector_type(8))) int          tdm_g4_t;

#ifndef __has_builtin
#define __has_builtin(x) 0
#endif
#if __has_builtin(__builtin_amdgcn_tensor_load_to_lds)
#define HAVE_TDM 1
#if defined(__clang_major__) && __clang_major__ >= 23
#define TDM_LOAD(g0, g1, g2, g3) \
    __builtin_amdgcn_tensor_load_to_lds((g0), (g1), (g2), (g3), (tdm_g4_t){0,0,0,0,0,0,0,0}, 0)
#else
#define TDM_LOAD(g0, g1, g2, g3) \
    __builtin_amdgcn_tensor_load_to_lds((g0), (g1), (g2), (g3), 0)
#endif
#else
#define HAVE_TDM 0
#endif

#define SHUF16(lo, hi) __builtin_shufflevector((lo), (hi), 0,1,2,3,4,5,6,7,8,9,10,11,12,13,14,15)

static constexpr int Bn = 32;       // batch
static constexpr int Tn = 2048;     // timesteps
static constexpr int Dn = 512;      // input dim
static constexpr int Hn = 512;      // hidden dim
static constexpr int Gn = 4 * Hn;   // 2048 gate columns
static constexpr int Mn = Bn * Tn;  // 65536 rows of the big GEMM

// workspace layout (bytes)
static constexpr size_t WS_BAR   = 0;                               // barrier counter
static constexpr size_t WS_H0    = 4096;                            // h ping  (bf16, 32x512)
static constexpr size_t WS_H1    = WS_H0 + 32768;                   // h pong
static constexpr size_t WS_WIT   = WS_H1 + 32768;                   // Wi^T bf16 [2048][512]
static constexpr size_t WS_XB    = WS_WIT + 2ull * 1024 * 1024;     // X bf16 [65536][512]
static constexpr size_t WS_XPROJ = WS_XB + 64ull * 1024 * 1024;     // x_proj f32 [65536][2048]
// total ≈ 606 MB

// ----- fragment loaders (contiguous k = 16h + s order, same for A and B) -----------
// A (MxK row-major): lanes 0-15 row m, K 0..15; lanes 16-31 K 16..31
__device__ __forceinline__ v16bf load_afrag(const __bf16* rowk, bool hiHalf) {
    const __bf16* p = rowk + (hiHalf ? 16 : 0);
    v8bf lo = *reinterpret_cast<const v8bf*>(p);
    v8bf hi = *reinterpret_cast<const v8bf*>(p + 8);
    return SHUF16(lo, hi);
}
// B (KxN, column n contiguous in K): lane holds 16 consecutive K of its column
__device__ __forceinline__ v16bf load_bfrag(const __bf16* colk, bool hiHalf) {
    const __bf16* p = colk + (hiHalf ? 16 : 0);
    v8bf lo = *reinterpret_cast<const v8bf*>(p);
    v8bf hi = *reinterpret_cast<const v8bf*>(p + 8);
    return SHUF16(lo, hi);
}

// ---------------------------------------------------------------------------
// conversion kernels
// ---------------------------------------------------------------------------
__global__ __launch_bounds__(256) void k_cvt_x(const float* __restrict__ X,
                                               unsigned short* __restrict__ Xb) {
    size_t i = ((size_t)blockIdx.x * 256 + threadIdx.x) * 4;
    if (i + 3 < (size_t)Mn * Dn) {
        float4 v = *reinterpret_cast<const float4*>(X + i);
        v4bf o = { (__bf16)v.x, (__bf16)v.y, (__bf16)v.z, (__bf16)v.w };
        *reinterpret_cast<v4bf*>(reinterpret_cast<__bf16*>(Xb) + i) = o;
    }
}

__global__ __launch_bounds__(256) void k_cvt_wiT(const float* __restrict__ Wi,
                                                 unsigned short* __restrict__ WiT) {
    size_t idx = (size_t)blockIdx.x * 256 + threadIdx.x;   // 512*2048 total
    if (idx < (size_t)Dn * Gn) {
        int n = (int)(idx & (Gn - 1));
        int k = (int)(idx >> 11);
        reinterpret_cast<__bf16*>(WiT)[(size_t)n * Dn + k] = (__bf16)Wi[(size_t)k * Gn + n];
    }
}

__global__ __launch_bounds__(256) void k_init_h(const float* __restrict__ h0,
                                                unsigned short* __restrict__ hb) {
    int idx = blockIdx.x * 256 + threadIdx.x;              // 32*512
    if (idx < Bn * Hn)
        reinterpret_cast<__bf16*>(hb)[idx] = (__bf16)h0[idx];
}

// ---------------------------------------------------------------------------
// Phase 1: x_proj = X @ Wi + b.  Grid 8192 WGs; WG tile 128x128; 8 waves;
// wave tile 64x32 -> 8 accumulators, 128 v_wmma per WG K-sweep.
// ---------------------------------------------------------------------------
__global__ __launch_bounds__(256) void k_gemm_xproj(const unsigned short* __restrict__ Xb_,
                                                    const unsigned short* __restrict__ WiT_,
                                                    const float* __restrict__ bias,
                                                    float* __restrict__ xproj) {
    const __bf16* Xb  = reinterpret_cast<const __bf16*>(Xb_);
    const __bf16* WiT = reinterpret_cast<const __bf16*>(WiT_);

    int nTileWG = blockIdx.x & 15;
    int mTileWG = blockIdx.x >> 4;
    int tid  = threadIdx.x;
    int w    = tid >> 5;
    int lane = tid & 31;
    int lh   = lane & 15;
    bool hiHalf = lane >= 16;

    int mBase = mTileWG * 128 + (w >> 2) * 64;   // 4 mtiles of 16
    int nBase = nTileWG * 128 + (w & 3) * 32;    // 2 ntiles of 16

    v8f acc[4][2] = {};

    for (int kk = 0; kk < 16; ++kk) {
        int kb = kk * 32;
        v16bf bf0 = load_bfrag(WiT + (size_t)(nBase + lh) * Dn + kb, hiHalf);
        v16bf bf1 = load_bfrag(WiT + (size_t)(nBase + 16 + lh) * Dn + kb, hiHalf);
#pragma unroll
        for (int m = 0; m < 4; ++m) {
            v16bf af = load_afrag(Xb + (size_t)(mBase + m * 16 + lh) * Dn + kb, hiHalf);
            acc[m][0] = __builtin_amdgcn_wmma_f32_16x16x32_bf16(
                false, af, false, bf0, (short)0, acc[m][0], false, false);
            acc[m][1] = __builtin_amdgcn_wmma_f32_16x16x32_bf16(
                false, af, false, bf1, (short)0, acc[m][1], false, false);
        }
    }

#pragma unroll
    for (int n = 0; n < 2; ++n) {
        float bv = bias[nBase + n * 16 + lh];
#pragma unroll
        for (int m = 0; m < 4; ++m) {
#pragma unroll
            for (int v = 0; v < 8; ++v) {
                int row = mBase + m * 16 + v + (hiHalf ? 8 : 0);
                xproj[(size_t)row * Gn + nBase + n * 16 + lh] = acc[m][n][v] + bv;
            }
        }
    }
}

// ---------------------------------------------------------------------------
// Phase 2: persistent scan. 16 WGs x 256 threads. WG owns 32 hidden units
// (= 128 z-columns across the 4 gates). Wave w: gate = w>>1, 16 units.
// Wh slice lives in 16 VGPR-resident B-fragments (128 VGPRs) for all steps.
// TDM prefetches next step's 32x128 x_proj tile into LDS (double-buffered),
// overlapping the DMA with this step's GEMM + device barrier.
// ---------------------------------------------------------------------------
__global__ __launch_bounds__(256) void k_lstm_scan(const float* __restrict__ xproj,
                                                   const float* __restrict__ Wh,
                                                   const float* __restrict__ c0,
                                                   unsigned short* __restrict__ h0buf_,
                                                   unsigned short* __restrict__ h1buf_,
                                                   float* __restrict__ out,
                                                   unsigned int* __restrict__ bar) {
    __bf16* hbufs[2] = { reinterpret_cast<__bf16*>(h0buf_), reinterpret_cast<__bf16*>(h1buf_) };

    int j0   = blockIdx.x * 32;          // hidden-unit slice of this WG
    int tid  = threadIdx.x;
    int w    = tid >> 5;
    int lane = tid & 31;
    int lh   = lane & 15;
    bool hiHalf = lane >= 16;

    int gate = w >> 1;                           // 0..3
    int ucol = (w & 1) * 16 + lh;                // unit within WG slice (0..31)
    int zcolGlobal = gate * Hn + j0 + ucol;      // column in [0,2048)
    int zlocal = gate * 32 + ucol;               // column in LDS tile

    // ---- gather this wave's Wh slice into persistent B-fragments (once) ----
    v16bf bfrag[16];
#pragma unroll
    for (int kk = 0; kk < 16; ++kk) {
        int kb = kk * 32 + (hiHalf ? 16 : 0);
        v16bf f;
#pragma unroll
        for (int e = 0; e < 16; ++e)
            f[e] = (__bf16)Wh[(size_t)(kb + e) * Gn + zcolGlobal];
        bfrag[kk] = f;
    }

    // ---- persistent cell state: 4 (batch, unit) pairs per thread ----
    float c[4];
    int bIdx[4], uIdx[4];
#pragma unroll
    for (int p = 0; p < 4; ++p) {
        int pair = p * 256 + tid;        // 1024 pairs = 32 batches x 32 units
        bIdx[p] = pair >> 5;
        uIdx[p] = pair & 31;
        c[p] = c0[bIdx[p] * Hn + j0 + uIdx[p]];
    }

    __shared__ float zlds[Bn][128];

#if HAVE_TDM
    __shared__ float xt[2][Bn][128];     // double-buffered x_proj step tile (32 KB)

    // Static TDM descriptor pieces (ISA §8.3-8.5): 4B data, 3D tile
    // dim0 = 32 contiguous cols, dim1 = 4 gates (stride 512),
    // dim2 = 32 batches (stride T*4H). LDS fill order == xt[b][gate*32+u].
    const unsigned long long s1 = (unsigned long long)Tn * Gn;  // dim1_stride (elems)
    tdm_g1_t g1;
    g1[0] = (int)(2u << 16);                                    // data_size = 4B
    g1[1] = (int)((512u & 0xFFFFu) << 16);                      // tensor_dim0 lo16
    g1[2] = (int)(((512u >> 16) & 0xFFFFu) | (4u << 16));       // dim0 hi16 | tensor_dim1 lo16
    g1[3] = (int)(32u << 16);                                   // dim1 hi16 | tile_dim0 = 32
    g1[4] = (int)(4u | (32u << 16));                            // tile_dim1 = 4, tile_dim2 = 32
    g1[5] = (int)512;                                           // tensor_dim0_stride lo32
    g1[6] = (int)((unsigned)(s1 & 0xFFFFu) << 16);              // stride0 hi16 | stride1 lo16
    g1[7] = (int)((s1 >> 16) & 0xFFFFFFFFull);                  // stride1 [47:16]
    tdm_g2_t g2 = { 32, 0, 0, 0 };                              // tensor_dim2 = 32
    tdm_g2_t g3 = { 0, 0, 0, 0 };
    unsigned ldsOff[2] = { (unsigned)(uintptr_t)&xt[0][0][0],
                           (unsigned)(uintptr_t)&xt[1][0][0] };

    auto tdm_issue = [&](int tt, int buf) {
        unsigned long long ga =
            (unsigned long long)(uintptr_t)(xproj + (size_t)tt * Gn + j0);
        tdm_g0_t g0;
        g0[0] = 1u;                                             // count = 1
        g0[1] = ldsOff[buf];                                    // lds_addr
        g0[2] = (unsigned)(ga & 0xFFFFFFFFull);                 // global_addr lo
        g0[3] = (unsigned)((ga >> 32) & 0x1FFFFFFull) | 0x80000000u;  // hi | type=2
        TDM_LOAD(g0, g1, g2, g3);
    };

    if (w == 0) tdm_issue(0, 0);         // prologue: tile for t = 0
#endif

    for (int t = 0; t < Tn; ++t) {
        const __bf16* hrd = hbufs[t & 1];
        __bf16*       hwr = hbufs[(t + 1) & 1];

#if HAVE_TDM
        if (w == 0 && t + 1 < Tn) tdm_issue(t + 1, (t + 1) & 1);
        v8f acc[2] = {};                 // x_proj added later from LDS
#else
        // accumulator init from x_proj (bias already folded in)
        v8f acc[2];
#pragma unroll
        for (int m = 0; m < 2; ++m) {
#pragma unroll
            for (int v = 0; v < 8; ++v) {
                int bRow = m * 16 + v + (hiHalf ? 8 : 0);
                acc[m][v] = xproj[((size_t)bRow * Tn + t) * Gn + zcolGlobal];
                if (t + 1 < Tn)
                    __builtin_prefetch(&xproj[((size_t)bRow * Tn + t + 1) * Gn + zcolGlobal], 0, 1);
            }
        }
#endif

        // z += h @ Wh   (A from 32KB L2-hot h buffer, B from VGPRs)
#pragma unroll
        for (int kk = 0; kk < 16; ++kk) {
#pragma unroll
            for (int m = 0; m < 2; ++m) {
                v16bf af = load_afrag(hrd + (size_t)(m * 16 + lh) * Hn + kk * 32, hiHalf);
                acc[m] = __builtin_amdgcn_wmma_f32_16x16x32_bf16(
                    false, af, false, bfrag[kk], (short)0, acc[m], false, false);
            }
        }

        // publish z tile to LDS
#pragma unroll
        for (int m = 0; m < 2; ++m)
#pragma unroll
            for (int v = 0; v < 8; ++v) {
                int bRow = m * 16 + v + (hiHalf ? 8 : 0);
                zlds[bRow][zlocal] = acc[m][v];
            }

#if HAVE_TDM
        if (w == 0) {                    // in-order TENSORcnt: <=1 => tile t done
            if (t + 1 < Tn) __builtin_amdgcn_s_wait_tensorcnt(1);
            else            __builtin_amdgcn_s_wait_tensorcnt(0);
        }
#endif
        __syncthreads();

        // gates: i, f, g, o  ->  c, h
#pragma unroll
        for (int p = 0; p < 4; ++p) {
            int b = bIdx[p], u = uIdx[p];
#if HAVE_TDM
            const float* xb = &xt[t & 1][b][0];
            float zi = zlds[b][u]      + xb[u];
            float zf = zlds[b][32 + u] + xb[32 + u];
            float zg = zlds[b][64 + u] + xb[64 + u];
            float zo = zlds[b][96 + u] + xb[96 + u];
#else
            float zi = zlds[b][u];
            float zf = zlds[b][32 + u];
            float zg = zlds[b][64 + u];
            float zo = zlds[b][96 + u];
#endif
            float si = 1.0f / (1.0f + __expf(-zi));
            float sf = 1.0f / (1.0f + __expf(-zf));
            float so = 1.0f / (1.0f + __expf(-zo));
            float cn = sf * c[p] + si * tanhf(zg);
            c[p] = cn;
            float h = so * tanhf(cn);
            out[((size_t)b * Tn + t) * Hn + j0 + u] = h;
            hwr[(size_t)b * Hn + j0 + u] = (__bf16)h;
        }
        __syncthreads();

        // device-wide step barrier (monotonic counter: no reset race)
        __threadfence();
        if (tid == 0) {
            __hip_atomic_fetch_add(bar, 1u, __ATOMIC_ACQ_REL, __HIP_MEMORY_SCOPE_AGENT);
            unsigned target = 16u * (unsigned)(t + 1);
            while (__hip_atomic_load(bar, __ATOMIC_ACQUIRE, __HIP_MEMORY_SCOPE_AGENT) < target)
                __builtin_amdgcn_s_sleep(1);
        }
        __syncthreads();
    }
}

// ---------------------------------------------------------------------------
extern "C" void kernel_launch(void* const* d_in, const int* in_sizes, int n_in,
                              void* d_out, int out_size, void* d_ws, size_t ws_size,
                              hipStream_t stream) {
    (void)in_sizes; (void)n_in; (void)out_size; (void)ws_size;
    const float* X    = (const float*)d_in[0];   // [B,T,D]
    // d_in[1] = input_paddings (all valid; unused, matching reference)
    const float* c0   = (const float*)d_in[2];   // [B,H]
    const float* h0   = (const float*)d_in[3];   // [B,H]
    const float* Wi   = (const float*)d_in[4];   // [D,4H]
    const float* Wh   = (const float*)d_in[5];   // [H,4H]
    const float* bias = (const float*)d_in[6];   // [4H]
    float* out = (float*)d_out;

    char* ws = (char*)d_ws;
    unsigned int*   bar   = (unsigned int*)(ws + WS_BAR);
    unsigned short* h0buf = (unsigned short*)(ws + WS_H0);
    unsigned short* h1buf = (unsigned short*)(ws + WS_H1);
    unsigned short* WiT   = (unsigned short*)(ws + WS_WIT);
    unsigned short* Xb    = (unsigned short*)(ws + WS_XB);
    float*          xproj = (float*)(ws + WS_XPROJ);

    hipMemsetAsync(bar, 0, 256, stream);                       // graph-safe

    k_cvt_x  <<<(Mn * Dn) / (256 * 4), 256, 0, stream>>>(X, Xb);
    k_cvt_wiT<<<(Dn * Gn) / 256,       256, 0, stream>>>(Wi, WiT);
    k_init_h <<<(Bn * Hn) / 256,       256, 0, stream>>>(h0, h0buf);

    k_gemm_xproj<<<(Mn / 128) * (Gn / 128), 256, 0, stream>>>(Xb, WiT, bias, xproj);

    k_lstm_scan<<<16, 256, 0, stream>>>(xproj, Wh, c0, h0buf, h1buf, out, bar);
}
